// GAT_82094004896437
// MI455X (gfx1250) — matrix operand order repaired and checked by reading, not compiled, not run on hardware
//
#include <hip/hip_runtime.h>
#include <hip/hip_bf16.h>
#include <math.h>

// ---- problem constants (match reference) ----
#define NN   65536      // B*V nodes
#define FEAT 8
#define HH1  2
#define CC1  32
#define HC1  64         // H1*C1
#define DD2  72         // H1*C1 + F
#define CC2  32
#define EE   1048576
#define ET   (EE + NN)  // edges incl. self-loops = 1114112
#define K1   65536      // FFN1 K = V*C2
#define N1   128
#define N2   16384

typedef __attribute__((ext_vector_type(16))) _Float16 v16h;
typedef __attribute__((ext_vector_type(8)))  _Float16 v8h;
typedef __attribute__((ext_vector_type(8)))  float    v8f;

// ---- helpers ----
__device__ __forceinline__ unsigned fkey(float f) {   // order-preserving float->uint
  unsigned u = __float_as_uint(f);
  return (u & 0x80000000u) ? ~u : (u | 0x80000000u);
}
__device__ __forceinline__ float funkey(unsigned k) {
  unsigned u = (k & 0x80000000u) ? (k & 0x7fffffffu) : ~k;
  return __uint_as_float(u);
}
__device__ __forceinline__ float lrelu(float v) { return v > 0.f ? v : 0.2f * v; }
__device__ __forceinline__ float eluf(float v)  { return v > 0.f ? v : expm1f(v); }

__device__ __forceinline__ v16h cat16(v8h lo, v8h hi) {
  v16h r;
#pragma unroll
  for (int i = 0; i < 8; ++i) { r[i] = lo[i]; r[i + 8] = hi[i]; }
  return r;
}

// ================= node projections =================
__global__ void proj1_kernel(const float* __restrict__ x,
                             const float* __restrict__ Wl, const float* __restrict__ bl,
                             const float* __restrict__ Wr, const float* __restrict__ br,
                             float* __restrict__ xl, float* __restrict__ xr) {
  __shared__ float sWl[FEAT * HC1], sWr[FEAT * HC1], sbl[HC1], sbr[HC1];
  for (int i = threadIdx.x; i < FEAT * HC1; i += blockDim.x) { sWl[i] = Wl[i]; sWr[i] = Wr[i]; }
  if (threadIdx.x < HC1) { sbl[threadIdx.x] = bl[threadIdx.x]; sbr[threadIdx.x] = br[threadIdx.x]; }
  __syncthreads();
  int n = blockIdx.x * blockDim.x + threadIdx.x;
  if (n >= NN) return;
  float xv[FEAT];
#pragma unroll
  for (int k = 0; k < FEAT; ++k) xv[k] = x[(size_t)n * FEAT + k];
#pragma unroll 4
  for (int c = 0; c < HC1; ++c) {
    float al = sbl[c], ar = sbr[c];
#pragma unroll
    for (int k = 0; k < FEAT; ++k) { al += xv[k] * sWl[k * HC1 + c]; ar += xv[k] * sWr[k * HC1 + c]; }
    xl[(size_t)n * HC1 + c] = al;
    xr[(size_t)n * HC1 + c] = ar;
  }
}

__global__ void proj2_kernel(const float* __restrict__ z,
                             const float* __restrict__ Wl, const float* __restrict__ bl,
                             const float* __restrict__ Wr, const float* __restrict__ br,
                             float* __restrict__ xl, float* __restrict__ xr) {
  __shared__ float sWl[DD2 * CC2], sWr[DD2 * CC2], sbl[CC2], sbr[CC2];
  for (int i = threadIdx.x; i < DD2 * CC2; i += blockDim.x) { sWl[i] = Wl[i]; sWr[i] = Wr[i]; }
  if (threadIdx.x < CC2) { sbl[threadIdx.x] = bl[threadIdx.x]; sbr[threadIdx.x] = br[threadIdx.x]; }
  __syncthreads();
  int n = blockIdx.x * blockDim.x + threadIdx.x;
  if (n >= NN) return;
  float zv[DD2];
#pragma unroll
  for (int k = 0; k < DD2; ++k) zv[k] = z[(size_t)n * DD2 + k];
#pragma unroll 4
  for (int c = 0; c < CC2; ++c) {
    float al = sbl[c], ar = sbr[c];
#pragma unroll 8
    for (int k = 0; k < DD2; ++k) { al += zv[k] * sWl[k * CC2 + c]; ar += zv[k] * sWr[k * CC2 + c]; }
    xl[(size_t)n * CC2 + c] = al;
    xr[(size_t)n * CC2 + c] = ar;
  }
}

// ================= GATv2 edge passes =================
template <int H, int C>
__global__ void edge_logit_max(const float* __restrict__ xl, const float* __restrict__ xr,
                               const int* __restrict__ src, const int* __restrict__ dst,
                               const float* __restrict__ att,
                               float* __restrict__ elog, unsigned* __restrict__ mkey,
                               int nE, int nReal) {
  __shared__ float satt[H * C];
  for (int i2 = threadIdx.x; i2 < H * C; i2 += blockDim.x) satt[i2] = att[i2];
  __syncthreads();
  int e = blockIdx.x * blockDim.x + threadIdx.x;
  if (e >= nE) return;
  int j, i;
  if (e < nReal) { j = src[e]; i = dst[e]; } else { j = i = e - nReal; }
  const float4* pl = (const float4*)(xl + (size_t)j * (H * C));
  const float4* pr = (const float4*)(xr + (size_t)i * (H * C));
#pragma unroll
  for (int h = 0; h < H; ++h) {
    float lg = 0.f;
#pragma unroll
    for (int q = 0; q < C / 4; ++q) {
      float4 a = pl[h * (C / 4) + q], b = pr[h * (C / 4) + q];
      lg += lrelu(a.x + b.x) * satt[h * C + 4 * q + 0];
      lg += lrelu(a.y + b.y) * satt[h * C + 4 * q + 1];
      lg += lrelu(a.z + b.z) * satt[h * C + 4 * q + 2];
      lg += lrelu(a.w + b.w) * satt[h * C + 4 * q + 3];
    }
    elog[(size_t)e * H + h] = lg;
    atomicMax(&mkey[(size_t)i * H + h], fkey(lg));
  }
}

template <int H>
__global__ void edge_exp_sum(const int* __restrict__ dst, const unsigned* __restrict__ mkey,
                             float* __restrict__ elog, float* __restrict__ ssum,
                             int nE, int nReal) {
  int e = blockIdx.x * blockDim.x + threadIdx.x;
  if (e >= nE) return;
  int i = (e < nReal) ? dst[e] : (e - nReal);
#pragma unroll
  for (int h = 0; h < H; ++h) {
    float m  = funkey(mkey[(size_t)i * H + h]);
    float pv = expf(elog[(size_t)e * H + h] - m);
    elog[(size_t)e * H + h] = pv;
    atomicAdd(&ssum[(size_t)i * H + h], pv);
  }
}

template <int H, int C>
__global__ void edge_agg(const float* __restrict__ xl,
                         const int* __restrict__ src, const int* __restrict__ dst,
                         const float* __restrict__ p, const float* __restrict__ ssum,
                         float* __restrict__ out, int nE, int nReal) {
  int t = blockIdx.x * blockDim.x + threadIdx.x;
  if (t >= nE * H) return;
  int e = t / H, h = t - e * H;
  int j, i;
  if (e < nReal) { j = src[e]; i = dst[e]; } else { j = i = e - nReal; }
  float alpha = p[(size_t)e * H + h] / ssum[(size_t)i * H + h];
  const float* v = xl + (size_t)j * (H * C) + h * C;
  float* ov = out + (size_t)i * (H * C) + h * C;
#pragma unroll
  for (int c2 = 0; c2 < C; ++c2) atomicAdd(&ov[c2], v[c2] * alpha);
}

// ================= BatchNorm =================
template <int CH>
__global__ void bn_stats(const float* __restrict__ y, const float* __restrict__ g,
                         const float* __restrict__ be,
                         float* __restrict__ scale, float* __restrict__ shift, int n) {
  int c = blockIdx.x, t = threadIdx.x;
  float s = 0.f, s2 = 0.f;
  for (int r = t; r < n; r += blockDim.x) {
    float v = y[(size_t)r * CH + c];
    s += v; s2 += v * v;
  }
  __shared__ float sh[256], sh2[256];
  sh[t] = s; sh2[t] = s2; __syncthreads();
  for (int d = 128; d > 0; d >>= 1) {
    if (t < d) { sh[t] += sh[t + d]; sh2[t] += sh2[t + d]; }
    __syncthreads();
  }
  if (t == 0) {
    float mu = sh[0] / n;
    float var = sh2[0] / n - mu * mu;
    float sc = g[c] * rsqrtf(var + 1e-5f);
    scale[c] = sc;
    shift[c] = be[c] - mu * sc;
  }
}

__global__ void bn1_apply_concat(const float* __restrict__ y, const float* __restrict__ scale,
                                 const float* __restrict__ shift, const float* __restrict__ x,
                                 float* __restrict__ z) {
  __shared__ float ssc[HC1], ssh[HC1];
  if (threadIdx.x < HC1) { ssc[threadIdx.x] = scale[threadIdx.x]; ssh[threadIdx.x] = shift[threadIdx.x]; }
  __syncthreads();
  int n = blockIdx.x * blockDim.x + threadIdx.x;
  if (n >= NN) return;
#pragma unroll 8
  for (int c = 0; c < HC1; ++c)
    z[(size_t)n * DD2 + c] = eluf(y[(size_t)n * HC1 + c] * ssc[c] + ssh[c]);
#pragma unroll
  for (int k = 0; k < FEAT; ++k)
    z[(size_t)n * DD2 + HC1 + k] = x[(size_t)n * FEAT + k];
}

// BN2 + ELU, producing f16 A-operand for FFN1 (vectorized b128 stores)
__global__ void bn2_apply_h(const float* __restrict__ y, const float* __restrict__ scale,
                            const float* __restrict__ shift, _Float16* __restrict__ y2h) {
  __shared__ float ssc[CC2], ssh[CC2];
  if (threadIdx.x < CC2) { ssc[threadIdx.x] = scale[threadIdx.x]; ssh[threadIdx.x] = shift[threadIdx.x]; }
  __syncthreads();
  int n = blockIdx.x * blockDim.x + threadIdx.x;
  if (n >= NN) return;
#pragma unroll
  for (int b = 0; b < CC2 / 8; ++b) {
    v8h v;
#pragma unroll
    for (int q = 0; q < 8; ++q) {
      int c = b * 8 + q;
      v[q] = (_Float16)eluf(y[(size_t)n * CC2 + c] * ssc[c] + ssh[c]);
    }
    *(v8h*)(y2h + (size_t)n * CC2 + b * 8) = v;
  }
}

// ================= FFN operand packing (f32 -> f16, B transposed) =================
// Wf1 [K1,128] f32  ->  Wf1T [128][K1] f16  (column-contiguous B for WMMA)
__global__ void pack_wf1T(const float* __restrict__ Wf1, _Float16* __restrict__ Wt) {
  int t = blockIdx.x * blockDim.x + threadIdx.x;   // N1 * K1/8 = 1,048,576
  int n = t >> 13;                                 // / (K1/8)
  int k = (t & 8191) * 8;
  v8h v;
#pragma unroll
  for (int j = 0; j < 8; ++j) v[j] = (_Float16)Wf1[(size_t)(k + j) * N1 + n];
  *(v8h*)(Wt + (size_t)n * K1 + k) = v;
}

// Wf2 [128,N2] f32  ->  Wf2T [N2][128] f16
__global__ void pack_wf2T(const float* __restrict__ Wf2, _Float16* __restrict__ Wt) {
  int t = blockIdx.x * blockDim.x + threadIdx.x;   // N2 * 16 = 262,144
  int n = t >> 4;
  int k = (t & 15) * 8;
  v8h v;
#pragma unroll
  for (int j = 0; j < 8; ++j) v[j] = (_Float16)Wf2[(size_t)(k + j) * N2 + n];
  *(v8h*)(Wt + (size_t)n * 128 + k) = v;
}

// ================= FFN with WMMA =================
// WMMA f16 operand layouts (wave32), CDNA5 ISA 7.12.2:
//  A (16x32 MxK): lane group g holds K blocks {g*8..g*8+7} and {16+g*8..16+g*8+7}  -> 2x b128
//  B (32x16 KxN): lane = column, K blocks {g*8..g*8+7} and {16+g*8..16+g*8+7} (transposed storage) -> 2x b128
#define KCH 64
__global__ void ffn1_wmma(const _Float16* __restrict__ y2h, const _Float16* __restrict__ Wt,
                          float* __restrict__ acc) {
  int wave = threadIdx.x >> 5, lane = threadIdx.x & 31;
  int mt = blockIdx.x & 1, kc = blockIdx.x >> 1;
  int n0 = wave * 16, m0 = mt * 16;
  int g = lane >> 4, rr = lane & 15;
  const _Float16* arow = y2h + (size_t)(m0 + rr) * K1;
  const _Float16* bcol = Wt + (size_t)(n0 + rr) * K1;
  v8f c = {};
  int kbase = kc * (K1 / KCH);
#pragma unroll 2
  for (int s = 0; s < (K1 / KCH) / 32; ++s) {   // 32 wmma steps per wave
    int k0 = kbase + s * 32;
    if (k0 + 256 < K1) __builtin_prefetch(bcol + k0 + 256, 0, 1);
    v8h alo = *(const v8h*)(arow + k0 + g * 8);
    v8h ahi = *(const v8h*)(arow + k0 + 16 + g * 8);
    v8h blo = *(const v8h*)(bcol + k0 + g * 8);
    v8h bhi = *(const v8h*)(bcol + k0 + 16 + g * 8);
    c = __builtin_amdgcn_wmma_f32_16x16x32_f16(false, cat16(alo, ahi),
                                               false, cat16(blo, bhi),
                                               (short)0, c, false, false);
  }
#pragma unroll
  for (int j = 0; j < 8; ++j)
    atomicAdd(&acc[(size_t)(m0 + j + 8 * g) * N1 + n0 + rr], c[j]);
}

__global__ void ffn1_fin(const float* __restrict__ acc, const float* __restrict__ bf1,
                         _Float16* __restrict__ h16) {
  int i = blockIdx.x * blockDim.x + threadIdx.x;
  if (i < 32 * N1) {
    float v = acc[i] + bf1[i & (N1 - 1)];
    h16[i] = (_Float16)(v > 0.f ? v : 0.f);
  }
}

// out[32,N2] = relu(h16[32,128] @ Wf2T^T + bf2)
__global__ void ffn2_wmma(const _Float16* __restrict__ h16, const _Float16* __restrict__ Wt,
                          const float* __restrict__ bf2, float* __restrict__ out) {
  int wave = threadIdx.x >> 5, lane = threadIdx.x & 31;
  int tid = blockIdx.x * 8 + wave;           // 0..2047 tiles
  int nt = tid & 1023, mt = tid >> 10;
  int n0 = nt * 16, m0 = mt * 16;
  int g = lane >> 4, rr = lane & 15;
  const _Float16* arow = h16 + (size_t)(m0 + rr) * 128;
  const _Float16* bcol = Wt + (size_t)(n0 + rr) * 128;
  v8f c = {};
#pragma unroll
  for (int s = 0; s < 4; ++s) {
    int k0 = s * 32;
    v8h alo = *(const v8h*)(arow + k0 + g * 8);
    v8h ahi = *(const v8h*)(arow + k0 + 16 + g * 8);
    v8h blo = *(const v8h*)(bcol + k0 + g * 8);
    v8h bhi = *(const v8h*)(bcol + k0 + 16 + g * 8);
    c = __builtin_amdgcn_wmma_f32_16x16x32_f16(false, cat16(alo, ahi),
                                               false, cat16(blo, bhi),
                                               (short)0, c, false, false);
  }
#pragma unroll
  for (int j = 0; j < 8; ++j) {
    int m = m0 + j + 8 * g, n = n0 + rr;
    float v = c[j] + bf2[n];
    out[(size_t)m * N2 + n] = v > 0.f ? v : 0.f;
  }
}

// ================= launcher =================
extern "C" void kernel_launch(void* const* d_in, const int* in_sizes, int n_in,
                              void* d_out, int out_size, void* d_ws, size_t ws_size,
                              hipStream_t stream) {
  (void)in_sizes; (void)n_in; (void)out_size; (void)ws_size;
  const float* x    = (const float*)d_in[0];
  const int*   ei   = (const int*)d_in[1];
  const float* Wl1  = (const float*)d_in[2];
  const float* bl1  = (const float*)d_in[3];
  const float* Wr1  = (const float*)d_in[4];
  const float* br1  = (const float*)d_in[5];
  const float* att1 = (const float*)d_in[6];
  // d_in[7] = bias1 : per-channel add immediately before BatchNorm -> cancels exactly
  const float* g1   = (const float*)d_in[8];
  const float* be1  = (const float*)d_in[9];
  const float* Wl2  = (const float*)d_in[10];
  const float* bl2  = (const float*)d_in[11];
  const float* Wr2  = (const float*)d_in[12];
  const float* br2  = (const float*)d_in[13];
  const float* att2 = (const float*)d_in[14];
  // d_in[15] = bias2 : cancels in BN as well
  const float* g2   = (const float*)d_in[16];
  const float* be2  = (const float*)d_in[17];
  const float* Wf1  = (const float*)d_in[18];
  const float* bf1  = (const float*)d_in[19];
  const float* Wf2  = (const float*)d_in[20];
  const float* bf2  = (const float*)d_in[21];
  const int* srcE = ei;
  const int* dstE = ei + EE;

  // workspace arena (~130 MB)
  float* W = (float*)d_ws;
  size_t o = 0;
  float* xl1 = W + o; o += (size_t)NN * HC1;
  float* xr1 = W + o; o += (size_t)NN * HC1;
  float* y1  = W + o; o += (size_t)NN * HC1;
  float* z   = W + o; o += (size_t)NN * DD2;
  float* xl2 = W + o; o += (size_t)NN * CC2;
  float* xr2 = W + o; o += (size_t)NN * CC2;
  float* y2  = W + o; o += (size_t)NN * CC2;
  float* ebuf = W + o; o += (size_t)ET * HH1;
  float* msb  = W + o; o += (size_t)NN * 6;   // m1(2N) s1(2N) m2(N) s2(N), zeroed together
  unsigned* m1 = (unsigned*)msb;
  float*    s1 = msb + 2 * (size_t)NN;
  unsigned* m2 = (unsigned*)(msb + 4 * (size_t)NN);
  float*    s2 = msb + 5 * (size_t)NN;
  float* sc1 = W + o; o += 64;
  float* sh1 = W + o; o += 64;
  float* sc2 = W + o; o += 32;
  float* sh2 = W + o; o += 32;
  float* acc = W + o; o += 32 * N1;
  _Float16* h16  = (_Float16*)(W + o); o += 32 * N1 / 2 + 8;
  _Float16* y2h  = (_Float16*)(W + o); o += (size_t)NN * CC2 / 2;
  _Float16* wf1T = (_Float16*)(W + o); o += (size_t)N1 * K1 / 2;
  _Float16* wf2T = (_Float16*)(W + o); o += (size_t)N2 * 128 / 2;

  hipMemsetAsync(y1,  0, (size_t)NN * HC1 * sizeof(float), stream);
  hipMemsetAsync(y2,  0, (size_t)NN * CC2 * sizeof(float), stream);
  hipMemsetAsync(msb, 0, (size_t)NN * 6 * sizeof(float), stream);
  hipMemsetAsync(acc, 0, (size_t)32 * N1 * sizeof(float), stream);

  dim3 blk(256);
  // ---- FFN weight packing (independent; overlaps the graph front-end) ----
  pack_wf1T<<<(N1 * (K1 / 8)) / 256, blk, 0, stream>>>(Wf1, wf1T);
  pack_wf2T<<<(N2 * 16) / 256, blk, 0, stream>>>(Wf2, wf2T);
  // ---- GAT layer 1 ----
  proj1_kernel<<<NN / 256, blk, 0, stream>>>(x, Wl1, bl1, Wr1, br1, xl1, xr1);
  edge_logit_max<HH1, CC1><<<ET / 256, blk, 0, stream>>>(xl1, xr1, srcE, dstE, att1, ebuf, m1, ET, EE);
  edge_exp_sum<HH1><<<ET / 256, blk, 0, stream>>>(dstE, m1, ebuf, s1, ET, EE);
  edge_agg<HH1, CC1><<<(ET * HH1) / 256, blk, 0, stream>>>(xl1, srcE, dstE, ebuf, s1, y1, ET, EE);
  bn_stats<HC1><<<HC1, blk, 0, stream>>>(y1, g1, be1, sc1, sh1, NN);
  bn1_apply_concat<<<NN / 256, blk, 0, stream>>>(y1, sc1, sh1, x, z);
  // ---- GAT layer 2 ----
  proj2_kernel<<<NN / 256, blk, 0, stream>>>(z, Wl2, bl2, Wr2, br2, xl2, xr2);
  edge_logit_max<1, CC2><<<ET / 256, blk, 0, stream>>>(xl2, xr2, srcE, dstE, att2, ebuf, m2, ET, EE);
  edge_exp_sum<1><<<ET / 256, blk, 0, stream>>>(dstE, m2, ebuf, s2, ET, EE);
  edge_agg<1, CC2><<<ET / 256, blk, 0, stream>>>(xl2, srcE, dstE, ebuf, s2, y2, ET, EE);
  bn_stats<CC2><<<CC2, blk, 0, stream>>>(y2, g2, be2, sc2, sh2, NN);
  bn2_apply_h<<<NN / 256, blk, 0, stream>>>(y2, sc2, sh2, y2h);
  // ---- FFN (WMMA, f16 operands pre-packed, f32 accumulate) ----
  ffn1_wmma<<<2 * KCH, blk, 0, stream>>>(y2h, wf1T, acc);
  ffn1_fin<<<16, blk, 0, stream>>>(acc, bf1, h16);
  ffn2_wmma<<<256, blk, 0, stream>>>(h16, wf2T, bf2, (float*)d_out);
}
